// StarformerNER_75960791597427
// MI455X (gfx1250) — compile-verified, matrix operand order checked by reading, not run on hardware
//
#include <hip/hip_runtime.h>

#define C_H  768
#define C_NL 12
#define C_HD 64
#define C_B  8
#define C_S  512
#define C_M  4096   /* B*S */
#define CPAD 32     /* padded row count for 8-row "center" matrices */
#define NEGBIG -1.0e12f

typedef __attribute__((ext_vector_type(16))) __bf16        bf16x16;
typedef __attribute__((ext_vector_type(8)))  float         f32x8;
typedef __attribute__((ext_vector_type(4)))  unsigned int  u32x4;

union Frag { bf16x16 v; u32x4 q[2]; };

__device__ __forceinline__ unsigned short f2bf(float f) {
  unsigned int u = __float_as_uint(f);
  u = (u + 0x7FFFu + ((u >> 16) & 1u)) >> 16;   // round-to-nearest-even
  return (unsigned short)u;
}

// ---------------------------------------------------------------------------
// fp32 (KxN) weight  ->  bf16 transposed (Npad x Kpad), zero padded
// ---------------------------------------------------------------------------
__global__ void wt_conv(const float* __restrict__ W, unsigned short* __restrict__ Wt,
                        int K, int N, int Kpad, long long total) {
  long long idx = (long long)blockIdx.x * blockDim.x + threadIdx.x;
  if (idx >= total) return;
  long long n = idx / Kpad;
  int k = (int)(idx - n * Kpad);
  float v = (k < K && n < (long long)N) ? W[(long long)k * N + n] : 0.f;
  Wt[idx] = f2bf(v);
}

// fp32 activations (Mvalid x K) -> bf16 (Mtotal x Kpad), zero padded
__global__ void f2bf_pad(const float* __restrict__ A, unsigned short* __restrict__ O,
                         long long Mvalid, int K, int Kpad, long long total) {
  long long idx = (long long)blockIdx.x * blockDim.x + threadIdx.x;
  if (idx >= total) return;
  long long m = idx / Kpad;
  int k = (int)(idx - m * Kpad);
  float v = (m < Mvalid && k < K) ? A[m * (long long)K + k] : 0.f;
  O[idx] = f2bf(v);
}

__global__ void pad_bias(const float* __restrict__ b, float* __restrict__ o,
                         int n, int npad) {
  int i = threadIdx.x;
  if (i < npad) o[i] = (i < n) ? b[i] : 0.f;
}

// ---------------------------------------------------------------------------
// bf16 WMMA GEMM:  C[M,N] = alpha * (A[M,K(bf16)] x Wt[N,K(bf16)]^T) + bias[N]
// Wt is N x K row-major (weight transposed), so both A and B fragments are
// two contiguous 16B loads per lane, matching the ISA bf16 fragment layout
// (lanes 0-15: K=k..k+7 & k+16..k+23; lanes 16-31: the +8 offsets).
// One wave computes a 32x64 output tile: 2 A-frags x 4 B-frags ->
// 8 v_wmma_f32_16x16x32_bf16 per 32-wide K step (load:wmma = 1.5:1).
// __launch_bounds__(128, 1): let the allocator use the full VGPR budget so
// the 8 f32x8 accumulators + pipelined fragments stay in registers (no
// scratch spills); the 8 independent WMMAs per step self-hide latency.
// Requires M%32==0, N%64==0, K%32==0. Batched via blockIdx.z.
// ---------------------------------------------------------------------------
__global__ __launch_bounds__(128, 1)
void gemm_bf16(const unsigned short* __restrict__ A,
               const unsigned short* __restrict__ Wt,
               const float* __restrict__ bias,
               float* __restrict__ C,
               int M, int N, int K,
               long long sA, long long sW, long long sC,
               float alpha) {
  int wave = (int)((blockIdx.x * blockDim.x + threadIdx.x) >> 5);
  int lane = threadIdx.x & 31;
  int tilesM = M >> 5, tilesN = N >> 6;
  if (wave >= tilesM * tilesN) return;            // whole-wave exit: EXEC stays all-1s
  int tm = wave % tilesM;
  int tn = wave / tilesM;
  long long z = blockIdx.z;
  int koff = (lane < 16) ? 0 : 8;
  const unsigned short* A0 = A + z * sA + (long long)(tm * 32 + (lane & 15)) * K + koff;
  const unsigned short* A1 = A0 + (long long)16 * K;
  const unsigned short* B0 = Wt + z * sW + (long long)(tn * 64 + (lane & 15)) * K + koff;
  const unsigned short* B1 = B0 + (long long)16 * K;
  const unsigned short* B2 = B0 + (long long)32 * K;
  const unsigned short* B3 = B0 + (long long)48 * K;
  f32x8 acc00 = {0.f,0.f,0.f,0.f,0.f,0.f,0.f,0.f};
  f32x8 acc01 = acc00, acc02 = acc00, acc03 = acc00;
  f32x8 acc10 = acc00, acc11 = acc00, acc12 = acc00, acc13 = acc00;
  for (int k = 0; k < K; k += 32) {
    Frag a0, a1, b0, b1, b2, b3;
    a0.q[0] = *(const u32x4*)(A0 + k);  a0.q[1] = *(const u32x4*)(A0 + k + 16);
    a1.q[0] = *(const u32x4*)(A1 + k);  a1.q[1] = *(const u32x4*)(A1 + k + 16);
    b0.q[0] = *(const u32x4*)(B0 + k);  b0.q[1] = *(const u32x4*)(B0 + k + 16);
    b1.q[0] = *(const u32x4*)(B1 + k);  b1.q[1] = *(const u32x4*)(B1 + k + 16);
    b2.q[0] = *(const u32x4*)(B2 + k);  b2.q[1] = *(const u32x4*)(B2 + k + 16);
    b3.q[0] = *(const u32x4*)(B3 + k);  b3.q[1] = *(const u32x4*)(B3 + k + 16);
    if (k + 128 < K) {
      __builtin_prefetch((const void*)(A0 + k + 128), 0, 1);
      __builtin_prefetch((const void*)(A1 + k + 128), 0, 1);
      __builtin_prefetch((const void*)(B0 + k + 128), 0, 1);
      __builtin_prefetch((const void*)(B2 + k + 128), 0, 1);
    }
    acc00 = __builtin_amdgcn_wmma_f32_16x16x32_bf16(false, a0.v, false, b0.v, (short)0, acc00, false, false);
    acc01 = __builtin_amdgcn_wmma_f32_16x16x32_bf16(false, a0.v, false, b1.v, (short)0, acc01, false, false);
    acc02 = __builtin_amdgcn_wmma_f32_16x16x32_bf16(false, a0.v, false, b2.v, (short)0, acc02, false, false);
    acc03 = __builtin_amdgcn_wmma_f32_16x16x32_bf16(false, a0.v, false, b3.v, (short)0, acc03, false, false);
    acc10 = __builtin_amdgcn_wmma_f32_16x16x32_bf16(false, a1.v, false, b0.v, (short)0, acc10, false, false);
    acc11 = __builtin_amdgcn_wmma_f32_16x16x32_bf16(false, a1.v, false, b1.v, (short)0, acc11, false, false);
    acc12 = __builtin_amdgcn_wmma_f32_16x16x32_bf16(false, a1.v, false, b2.v, (short)0, acc12, false, false);
    acc13 = __builtin_amdgcn_wmma_f32_16x16x32_bf16(false, a1.v, false, b3.v, (short)0, acc13, false, false);
  }
  int nb = tn * 64 + (lane & 15);
  float bb0 = bias ? bias[nb]      : 0.f;
  float bb1 = bias ? bias[nb + 16] : 0.f;
  float bb2 = bias ? bias[nb + 32] : 0.f;
  float bb3 = bias ? bias[nb + 48] : 0.f;
  int m0 = tm * 32 + ((lane < 16) ? 0 : 8);
  float* Cb = C + z * sC;
#pragma unroll
  for (int j = 0; j < 8; ++j) {
    long long r0 = (long long)(m0 + j) * N;
    long long r1 = (long long)(m0 + 16 + j) * N;
    Cb[r0 + nb]      = acc00[j] * alpha + bb0;
    Cb[r0 + nb + 16] = acc01[j] * alpha + bb1;
    Cb[r0 + nb + 32] = acc02[j] * alpha + bb2;
    Cb[r0 + nb + 48] = acc03[j] * alpha + bb3;
    Cb[r1 + nb]      = acc10[j] * alpha + bb0;
    Cb[r1 + nb + 16] = acc11[j] * alpha + bb1;
    Cb[r1 + nb + 32] = acc12[j] * alpha + bb2;
    Cb[r1 + nb + 48] = acc13[j] * alpha + bb3;
  }
}

// ---------------------------------------------------------------------------
// center = hidden0.mean(axis=1)  ->  fp32 (CPAD x 768 padded) + bf16 copy
// ---------------------------------------------------------------------------
__global__ void col_mean(const float* __restrict__ h0, float* __restrict__ center,
                         unsigned short* __restrict__ cen_bf) {
  int idx = blockIdx.x * blockDim.x + threadIdx.x;
  if (idx >= CPAD * C_H) return;
  int row = idx / C_H, i = idx - row * C_H;
  float s = 0.f;
  if (row < C_B) {
    for (int ss = 0; ss < C_S; ++ss)
      s += h0[((long long)(row * C_S + ss)) * C_H + i];
    s *= (1.f / (float)C_S);
  }
  center[idx] = s;
  cen_bf[idx] = f2bf(s);
}

// ---------------------------------------------------------------------------
// sat attention: per (token, head), softmax over 5 gathered K rows.
// Keys/values of [before, cs, after, hidden0, center] are row-gathers of
// the projections of cs / hidden0 / center (shift commutes with matmul).
// ---------------------------------------------------------------------------
__global__ void sat_attn(const float* __restrict__ Qp, const float* __restrict__ KC,
                         const float* __restrict__ KH, const float* __restrict__ Kc,
                         const float* __restrict__ VC, const float* __restrict__ VH,
                         const float* __restrict__ Vc, float* __restrict__ ctx) {
  int id = blockIdx.x * blockDim.x + threadIdx.x;
  if (id >= C_M * C_NL) return;
  int t = id / C_NL, h = id - t * C_NL;
  int b = t >> 9, s = t & (C_S - 1);
  long long hb = (long long)h * C_HD;
  const float* q = Qp + (long long)t * C_H + hb;
  long long r0 = (long long)(b * C_S + ((s + 1) & (C_S - 1))) * C_H + hb;   // roll(-1)
  long long r1 = (long long)t * C_H + hb;                                   // cs
  long long r2 = (long long)(b * C_S + (s == 0 ? (C_S - 1) : 0)) * C_H + hb; // shift_fwd
  long long r4 = (long long)b * C_H + hb;                                   // center
  const float* kp[5] = { KC + r0, KC + r1, KC + r2, KH + r1, Kc + r4 };
  const float* vp[5] = { VC + r0, VC + r1, VC + r2, VH + r1, Vc + r4 };
  float sc[5];
  float mx = -3.4e38f;
  for (int j = 0; j < 5; ++j) {
    float d = 0.f;
    for (int i = 0; i < C_HD; ++i) d += q[i] * kp[j][i];
    sc[j] = d * 0.125f;
    mx = fmaxf(mx, sc[j]);
  }
  float ssum = 0.f;
  for (int j = 0; j < 5; ++j) { sc[j] = __expf(sc[j] - mx); ssum += sc[j]; }
  float inv = 1.f / ssum;
  float* op = ctx + r1;
  for (int i = 0; i < C_HD; ++i) {
    float a = 0.f;
    for (int j = 0; j < 5; ++j) a += sc[j] * vp[j][i];
    op[i] = a * inv;
  }
}

// ---------------------------------------------------------------------------
// x = relu(gemm_out + resid); layernorm -> outf (fp32) + outbf (bf16).
// rows >= validRows are zero-filled (padding rows).
// blockDim must be 256, H=768 (3 elems/thread).
// ---------------------------------------------------------------------------
__global__ void res_relu_ln(const float* __restrict__ x, const float* __restrict__ resid,
                            const float* __restrict__ w, const float* __restrict__ b,
                            float* __restrict__ outf, unsigned short* __restrict__ outbf,
                            int validRows) {
  __shared__ float red[256];
  int row = blockIdx.x, t = threadIdx.x;
  long long base = (long long)row * C_H;
  if (row >= validRows) {
    for (int e = 0; e < 3; ++e) { int i = t + 256 * e; outf[base + i] = 0.f; outbf[base + i] = 0; }
    return;
  }
  float v[3];
  float s = 0.f;
  for (int e = 0; e < 3; ++e) {
    int i = t + 256 * e;
    float val = x[base + i] + resid[base + i];
    val = val > 0.f ? val : 0.f;
    v[e] = val; s += val;
  }
  red[t] = s; __syncthreads();
  for (int o = 128; o > 0; o >>= 1) { if (t < o) red[t] += red[t + o]; __syncthreads(); }
  float mean = red[0] * (1.f / (float)C_H); __syncthreads();
  float d2 = 0.f;
  for (int e = 0; e < 3; ++e) { float d = v[e] - mean; d2 += d * d; }
  red[t] = d2; __syncthreads();
  for (int o = 128; o > 0; o >>= 1) { if (t < o) red[t] += red[t + o]; __syncthreads(); }
  float rstd = rsqrtf(red[0] * (1.f / (float)C_H) + 1e-12f);
  for (int e = 0; e < 3; ++e) {
    int i = t + 256 * e;
    float y = w[i] * (v[e] - mean) * rstd + b[i];
    outf[base + i] = y;
    outbf[base + i] = f2bf(y);
  }
}

// ---------------------------------------------------------------------------
// rel attention: one block per (batch, head), 513 keys = [center; cs].
// ---------------------------------------------------------------------------
__global__ void rel_attn(const float* __restrict__ qr, const float* __restrict__ Krc,
                         const float* __restrict__ Krs, const float* __restrict__ Vrc,
                         const float* __restrict__ Vrs, float* __restrict__ ctxr) {
  __shared__ float qsh[C_HD];
  __shared__ float sc[C_S + 1];
  __shared__ float red[256];
  int bh = blockIdx.x;
  int b = bh / C_NL, h = bh % C_NL;
  int t = threadIdx.x;
  long long hb = (long long)h * C_HD;
  if (t < C_HD) qsh[t] = qr[(long long)b * C_H + hb + t];
  __syncthreads();
  for (int n = t; n < C_S + 1; n += 256) {
    const float* kp = (n == 0) ? (Krc + (long long)b * C_H + hb)
                               : (Krs + (long long)(b * C_S + n - 1) * C_H + hb);
    float d = 0.f;
    for (int i = 0; i < C_HD; ++i) d += qsh[i] * kp[i];
    sc[n] = d * 0.125f;
  }
  __syncthreads();
  float lm = -3.4e38f;
  for (int n = t; n < C_S + 1; n += 256) lm = fmaxf(lm, sc[n]);
  red[t] = lm; __syncthreads();
  for (int o = 128; o > 0; o >>= 1) { if (t < o) red[t] = fmaxf(red[t], red[t + o]); __syncthreads(); }
  float gm = red[0]; __syncthreads();
  float ls = 0.f;
  for (int n = t; n < C_S + 1; n += 256) { float e = __expf(sc[n] - gm); sc[n] = e; ls += e; }
  red[t] = ls; __syncthreads();
  for (int o = 128; o > 0; o >>= 1) { if (t < o) red[t] += red[t + o]; __syncthreads(); }
  float inv = 1.f / red[0];
  if (t < C_HD) {
    float acc = 0.f;
    for (int n = 0; n <= C_S; ++n) {
      const float* vp = (n == 0) ? (Vrc + (long long)b * C_H + hb)
                                 : (Vrs + (long long)(b * C_S + n - 1) * C_H + hb);
      acc += sc[n] * vp[t];
    }
    ctxr[(long long)b * C_H + hb + t] = acc * inv;
  }
}

// ---------------------------------------------------------------------------
// De-interleave outputs[...,::2]/[...,1::2], apply RoPE, pack to bf16.
// ---------------------------------------------------------------------------
__global__ void rope_pack(const float* __restrict__ outputs,
                          unsigned short* __restrict__ qbf, unsigned short* __restrict__ kbf) {
  int idx = blockIdx.x * blockDim.x + threadIdx.x;
  if (idx >= C_M * (C_HD / 2)) return;
  int t = idx >> 5, i = idx & 31;
  int s = t & (C_S - 1);
  const float* o = outputs + (long long)t * 128;
  float q0 = o[4 * i],     q1 = o[4 * i + 2];
  float k0 = o[4 * i + 1], k1 = o[4 * i + 3];
  float invf = __expf(-(float)i * 0.28782314f);   // ln(10000)/32
  float sn, cs;
  __sincosf((float)s * invf, &sn, &cs);
  long long base = (long long)t * C_HD;
  qbf[base + 2 * i]     = f2bf(q0 * cs - q1 * sn);
  qbf[base + 2 * i + 1] = f2bf(q1 * cs + q0 * sn);
  kbf[base + 2 * i]     = f2bf(k0 * cs - k1 * sn);
  kbf[base + 2 * i + 1] = f2bf(k1 * cs + k0 * sn);
}

// ---------------------------------------------------------------------------
// final combine: qk (already /8 via alpha) + bias broadcasts + masks + tril.
// tailout has padded stride 64 (24 valid columns).
// ---------------------------------------------------------------------------
__global__ void combine(const float* __restrict__ qk, const float* __restrict__ tailout,
                        const float* __restrict__ mask, float* __restrict__ out,
                        long long total) {
  long long idx = (long long)blockIdx.x * blockDim.x + threadIdx.x;
  if (idx >= total) return;
  int ki = (int)(idx & (C_S - 1));
  long long r = idx >> 9;
  int qi = (int)(r & (C_S - 1)); r >>= 9;
  int h = (int)(r % C_NL);
  int b = (int)(r / C_NL);
  float v = qk[((long long)b * C_S + qi) * C_S + ki];
  v += tailout[((long long)(b * C_S + ki)) * 64 + 2 * h] * 0.5f
     + tailout[((long long)(b * C_S + qi)) * 64 + 2 * h + 1] * 0.5f;
  float m1 = mask[b * C_S + qi];
  v = v * m1 + NEGBIG * (1.f - m1);
  float m2 = mask[b * C_S + ki];
  v = v * m2 + NEGBIG * (1.f - m2);
  if (ki < qi) v -= 1.0e12f;
  out[idx] = v;
}

// ---------------------------------------------------------------------------
static inline void launch_gemm(const unsigned short* A, const unsigned short* Wt,
                               const float* bias, float* C, int M, int N, int K,
                               int batch, long long sA, long long sW, long long sC,
                               float alpha, hipStream_t st) {
  int tiles = (M >> 5) * (N >> 6);
  dim3 grid((unsigned)((tiles + 3) / 4), 1, (unsigned)batch);
  gemm_bf16<<<grid, dim3(128), 0, st>>>(A, Wt, bias, C, M, N, K, sA, sW, sC, alpha);
}

extern "C" void kernel_launch(void* const* d_in, const int* in_sizes, int n_in,
                              void* d_out, int out_size, void* d_ws, size_t ws_size,
                              hipStream_t stream) {
  (void)in_sizes; (void)n_in; (void)out_size; (void)ws_size;
  const float* bert    = (const float*)d_in[0];
  const float* amask   = (const float*)d_in[1];
  const float* dense_w = (const float*)d_in[2];
  const float* dense_b = (const float*)d_in[3];
  const float* sat_qw  = (const float*)d_in[4];
  const float* sat_qb  = (const float*)d_in[5];
  const float* sat_kw  = (const float*)d_in[6];
  const float* sat_kb  = (const float*)d_in[7];
  const float* sat_vw  = (const float*)d_in[8];
  const float* sat_vb  = (const float*)d_in[9];
  const float* sat_ow  = (const float*)d_in[10];
  const float* sat_ob  = (const float*)d_in[11];
  const float* rel_qw  = (const float*)d_in[12];
  const float* rel_qb  = (const float*)d_in[13];
  const float* rel_kw  = (const float*)d_in[14];
  const float* rel_kb  = (const float*)d_in[15];
  const float* rel_vw  = (const float*)d_in[16];
  const float* rel_vb  = (const float*)d_in[17];
  const float* rel_ow  = (const float*)d_in[18];
  const float* rel_ob  = (const float*)d_in[19];
  const float* lns_w   = (const float*)d_in[20];
  const float* lns_b   = (const float*)d_in[21];
  const float* lnr_w   = (const float*)d_in[22];
  const float* lnr_b   = (const float*)d_in[23];
  const float* head_w  = (const float*)d_in[24];
  const float* head_b  = (const float*)d_in[25];
  const float* tail_w  = (const float*)d_in[26];
  const float* tail_b  = (const float*)d_in[27];
  float* out = (float*)d_out;

  typedef unsigned short us;
  char* p = (char*)d_ws;
  auto alloc = [&](size_t bytes) -> char* {
    char* r = p;
    p += (bytes + 255) & ~(size_t)255;
    return r;
  };
  const long long MH = (long long)C_M * C_H;   // 3145728
  // bf16 weight (transposed) buffers
  us* bert_bf = (us*)alloc((size_t)C_M * 448 * 2);
  us* dense_t = (us*)alloc((size_t)768 * 448 * 2);
  us* wq_t = (us*)alloc((size_t)768 * 768 * 2);
  us* wk_t = (us*)alloc((size_t)768 * 768 * 2);
  us* wv_t = (us*)alloc((size_t)768 * 768 * 2);
  us* wo_t = (us*)alloc((size_t)768 * 768 * 2);
  us* rq_t = (us*)alloc((size_t)768 * 768 * 2);
  us* rk_t = (us*)alloc((size_t)768 * 768 * 2);
  us* rv_t = (us*)alloc((size_t)768 * 768 * 2);
  us* ro_t = (us*)alloc((size_t)768 * 768 * 2);
  us* head_t = (us*)alloc((size_t)128 * 768 * 2);
  us* tail_t = (us*)alloc((size_t)64 * 128 * 2);
  float* tail_b_pad = (float*)alloc(64 * 4);
  // big fp32 buffers (with aliasing across phases)
  float* hidden0 = (float*)alloc((size_t)MH * 4);
  float* cs      = (float*)alloc((size_t)MH * 4);
  float* Qp      = (float*)alloc((size_t)MH * 4);   // aliased as satout
  float* KC      = (float*)alloc((size_t)MH * 4);   // aliased as Krs
  float* KH      = (float*)alloc((size_t)MH * 4);
  float* VC      = (float*)alloc((size_t)MH * 4);   // aliased as Vrs
  float* VH      = (float*)alloc((size_t)MH * 4);
  float* ctx     = (float*)alloc((size_t)MH * 4);
  us* h0_bf  = (us*)alloc((size_t)MH * 2);
  us* cs_bf  = (us*)alloc((size_t)MH * 2);
  us* ctx_bf = (us*)alloc((size_t)MH * 2);
  // small (CPAD x 768 padded) buffers
  float* center = (float*)alloc((size_t)CPAD * 768 * 4);
  float* Kcen   = (float*)alloc((size_t)CPAD * 768 * 4);
  float* Vcen   = (float*)alloc((size_t)CPAD * 768 * 4);
  float* qr     = (float*)alloc((size_t)CPAD * 768 * 4);
  float* Krc    = (float*)alloc((size_t)CPAD * 768 * 4);
  float* Vrc    = (float*)alloc((size_t)CPAD * 768 * 4);
  float* ctxr   = (float*)alloc((size_t)CPAD * 768 * 4);
  float* relout = (float*)alloc((size_t)CPAD * 768 * 4);
  us* cen_bf  = (us*)alloc((size_t)CPAD * 768 * 2);
  us* ctxr_bf = (us*)alloc((size_t)CPAD * 768 * 2);
  // tail end buffers
  float* outputs = (float*)alloc((size_t)C_M * 128 * 4);
  us* out_bf     = (us*)alloc((size_t)C_M * 128 * 2);
  float* tailout = (float*)alloc((size_t)C_M * 64 * 4);
  us* qbf = (us*)alloc((size_t)C_M * C_HD * 2);
  us* kbf = (us*)alloc((size_t)C_M * C_HD * 2);
  float* qk = (float*)alloc((size_t)C_B * C_S * C_S * 4);

  // --- weight conversion (bf16, transposed, padded) ---
  auto wconv = [&](const float* W, us* Wt, int K, int N, int Kpad, int Npad) {
    long long tot = (long long)Npad * Kpad;
    wt_conv<<<(unsigned)((tot + 255) / 256), 256, 0, stream>>>(W, Wt, K, N, Kpad, tot);
  };
  wconv(dense_w, dense_t, 432, 768, 448, 768);
  wconv(sat_qw, wq_t, 768, 768, 768, 768);
  wconv(sat_kw, wk_t, 768, 768, 768, 768);
  wconv(sat_vw, wv_t, 768, 768, 768, 768);
  wconv(sat_ow, wo_t, 768, 768, 768, 768);
  wconv(rel_qw, rq_t, 768, 768, 768, 768);
  wconv(rel_kw, rk_t, 768, 768, 768, 768);
  wconv(rel_vw, rv_t, 768, 768, 768, 768);
  wconv(rel_ow, ro_t, 768, 768, 768, 768);
  wconv(head_w, head_t, 768, 128, 768, 128);
  wconv(tail_w, tail_t, 128, 24, 128, 64);
  pad_bias<<<1, 64, 0, stream>>>(tail_b, tail_b_pad, 24, 64);

  // --- dense: hidden0 = bert @ dense_w + dense_b ---
  {
    long long tot = (long long)C_M * 448;
    f2bf_pad<<<(unsigned)((tot + 255) / 256), 256, 0, stream>>>(bert, bert_bf, C_M, 432, 448, tot);
  }
  launch_gemm(bert_bf, dense_t, dense_b, hidden0, C_M, 768, 448, 1, 0, 0, 0, 1.f, stream);
  f2bf_pad<<<(unsigned)((MH + 255) / 256), 256, 0, stream>>>(hidden0, h0_bf, C_M, 768, 768, MH);
  hipMemcpyAsync(cs, hidden0, (size_t)MH * 4, hipMemcpyDeviceToDevice, stream);
  hipMemcpyAsync(cs_bf, h0_bf, (size_t)MH * 2, hipMemcpyDeviceToDevice, stream);
  col_mean<<<(CPAD * C_H + 255) / 256, 256, 0, stream>>>(hidden0, center, cen_bf);

  for (int it = 0; it < 2; ++it) {
    // sat projections (concats never materialized: shift commutes with matmul)
    launch_gemm(cs_bf,  wq_t, sat_qb, Qp,   C_M,  768, 768, 1, 0, 0, 0, 1.f, stream);
    launch_gemm(cs_bf,  wk_t, sat_kb, KC,   C_M,  768, 768, 1, 0, 0, 0, 1.f, stream);
    launch_gemm(h0_bf,  wk_t, sat_kb, KH,   C_M,  768, 768, 1, 0, 0, 0, 1.f, stream);
    launch_gemm(cs_bf,  wv_t, sat_vb, VC,   C_M,  768, 768, 1, 0, 0, 0, 1.f, stream);
    launch_gemm(h0_bf,  wv_t, sat_vb, VH,   C_M,  768, 768, 1, 0, 0, 0, 1.f, stream);
    launch_gemm(cen_bf, wk_t, sat_kb, Kcen, CPAD, 768, 768, 1, 0, 0, 0, 1.f, stream);
    launch_gemm(cen_bf, wv_t, sat_vb, Vcen, CPAD, 768, 768, 1, 0, 0, 0, 1.f, stream);
    sat_attn<<<192, 256, 0, stream>>>(Qp, KC, KH, Kcen, VC, VH, Vcen, ctx);
    f2bf_pad<<<(unsigned)((MH + 255) / 256), 256, 0, stream>>>(ctx, ctx_bf, C_M, 768, 768, MH);
    launch_gemm(ctx_bf, wo_t, sat_ob, Qp /*satout*/, C_M, 768, 768, 1, 0, 0, 0, 1.f, stream);
    res_relu_ln<<<C_M, 256, 0, stream>>>(Qp, cs, lns_w, lns_b, cs, cs_bf, C_M);
    // rel projections (Krs aliases KC, Vrs aliases VC)
    launch_gemm(cen_bf, rq_t, rel_qb, qr,  CPAD, 768, 768, 1, 0, 0, 0, 1.f, stream);
    launch_gemm(cen_bf, rk_t, rel_kb, Krc, CPAD, 768, 768, 1, 0, 0, 0, 1.f, stream);
    launch_gemm(cen_bf, rv_t, rel_vb, Vrc, CPAD, 768, 768, 1, 0, 0, 0, 1.f, stream);
    launch_gemm(cs_bf,  rk_t, rel_kb, KC,  C_M,  768, 768, 1, 0, 0, 0, 1.f, stream);
    launch_gemm(cs_bf,  rv_t, rel_vb, VC,  C_M,  768, 768, 1, 0, 0, 0, 1.f, stream);
    rel_attn<<<96, 256, 0, stream>>>(qr, Krc, KC, Vrc, VC, ctxr);
    f2bf_pad<<<(CPAD * C_H + 255) / 256, 256, 0, stream>>>(ctxr, ctxr_bf, C_B, 768, 768, (long long)CPAD * 768);
    launch_gemm(ctxr_bf, ro_t, rel_ob, relout, CPAD, 768, 768, 1, 0, 0, 0, 1.f, stream);
    res_relu_ln<<<CPAD, 256, 0, stream>>>(relout, center, lnr_w, lnr_b, center, cen_bf, C_B);
  }

  // head / tail / rope / logits / combine
  launch_gemm(cs_bf, head_t, head_b, outputs, C_M, 128, 768, 1, 0, 0, 0, 1.f, stream);
  {
    long long tot = (long long)C_M * 128;
    f2bf_pad<<<(unsigned)((tot + 255) / 256), 256, 0, stream>>>(outputs, out_bf, C_M, 128, 128, tot);
  }
  launch_gemm(out_bf, tail_t, tail_b_pad, tailout, C_M, 64, 128, 1, 0, 0, 0, 1.f, stream);
  rope_pack<<<512, 256, 0, stream>>>(outputs, qbf, kbf);
  launch_gemm(qbf, kbf, nullptr, qk, C_S, C_S, C_HD, C_B,
              (long long)C_S * C_HD, (long long)C_S * C_HD, (long long)C_S * C_S,
              0.125f, stream);
  {
    long long tot = (long long)C_B * C_NL * C_S * C_S;   // 25165824
    combine<<<(unsigned)((tot + 255) / 256), 256, 0, stream>>>(qk, tailout, amask, out, tot);
  }
}
